// CompressedSparseAttention_52424370815197
// MI455X (gfx1250) — matrix operand order repaired and checked by reading, not compiled, not run on hardware
//
#include <hip/hip_runtime.h>

// ---------------- problem constants ----------------
#define BB 2
#define TT 4096
#define DIM 2048
#define NH 16
#define HKV 4
#define HD 128
#define NREP 4
#define RR 16
#define NB 256          // TT / RR
#define TOPK 64
#define NCAT 3328       // 2048 q + 512 k + 512 v + 128 qi + 128 ki
#define MTOT (BB*TT)    // 8192

typedef __attribute__((ext_vector_type(16))) __bf16 v16bf;
typedef __attribute__((ext_vector_type(8)))  __bf16 v8bf;
typedef __attribute__((ext_vector_type(8)))  float  v8f;

union Frag16 { v16bf v; v8bf h[2]; };

__device__ __forceinline__ v8f wmma_bf16(const Frag16& a, const Frag16& b, v8f c) {
    return __builtin_amdgcn_wmma_f32_16x16x32_bf16(false, a.v, false, b.v,
                                                   (short)0, c, false, false);
}

// ---- CDNA5 async global->LDS copy (ASYNCcnt path), 16B per lane ----
// LDS operand = low 32 bits of the flat shared pointer (LDS aperture: addr[31:0]).
__device__ __forceinline__ void async_ld16(void* lds, const void* g) {
    unsigned l = (unsigned)(unsigned long long)lds;
    unsigned long long ga = (unsigned long long)g;
    asm volatile("global_load_async_to_lds_b128 %0, %1, off"
                 :: "v"(l), "v"(ga) : "memory");
}
__device__ __forceinline__ void wait_async0() {
    asm volatile("s_wait_asynccnt 0" ::: "memory");
}

// ---------------- elementwise convert / pack (weights pre-transposed, K-contiguous) ----------------
__global__ __launch_bounds__(256) void k_cvt_bf16(const float* __restrict__ src,
                                                  __bf16* __restrict__ dst, long n) {
    long i = (long)blockIdx.x * 256 + threadIdx.x;
    if (i < n) dst[i] = (__bf16)src[i];
}

// wcatT[n][k] = concat(wq,wk,wv,wiq,wik)[k][n]   -> [NCAT, DIM] bf16
__global__ __launch_bounds__(256) void k_pack_wcatT(
    const float* __restrict__ wq, const float* __restrict__ wk,
    const float* __restrict__ wv, const float* __restrict__ wiq,
    const float* __restrict__ wik, __bf16* __restrict__ wcatT) {
    long i = (long)blockIdx.x * 256 + threadIdx.x;
    if (i >= (long)DIM * NCAT) return;
    int k = (int)(i % DIM);
    int n = (int)(i / DIM);
    float v;
    if      (n < 2048) v = wq [(long)k * 2048 + n];
    else if (n < 2560) v = wk [(long)k * 512  + (n - 2048)];
    else if (n < 3072) v = wv [(long)k * 512  + (n - 2560)];
    else if (n < 3200) v = wiq[(long)k * 128  + (n - 3072)];
    else               v = wik[(long)k * 128  + (n - 3200)];
    wcatT[i] = (__bf16)v;
}

// woT[n][k] = wo[k][n]  -> [DIM, DIM] bf16
__global__ __launch_bounds__(256) void k_transpose_wo(const float* __restrict__ wo,
                                                      __bf16* __restrict__ woT) {
    long i = (long)blockIdx.x * 256 + threadIdx.x;
    if (i >= (long)DIM * DIM) return;
    int k = (int)(i % DIM);
    int n = (int)(i / DIM);
    woT[i] = (__bf16)wo[(long)k * DIM + n];
}

// ---------------- bf16 WMMA GEMM: C[M,N] = A[M,K] @ Bt[N,K]^T ----------------
// A row-major [M,K], Bt row-major [N,K] (B pre-transposed), C fp32 [M,N].
// M % 128 == 0, N % 256 == 0, K % 32 == 0.
// Double-buffered LDS, staged via async global->LDS DMA (no VGPR round-trip).
#define TM 128
#define TN 256
#define TK 32
#define LDT 40                        // 32 + 8 bf16 pad, keeps 16B alignment
#define SMSZ ((TM + TN) * LDT)        // one buffer: A rows [0,128), B rows [128,384)

__global__ __launch_bounds__(256) void k_gemm_bf16(
    const __bf16* __restrict__ A, const __bf16* __restrict__ Bt,
    float* __restrict__ C, int M, int N, int K) {
    __shared__ __bf16 sm[2][SMSZ];
    const int bn = blockIdx.x, bm = blockIdx.y;
    const int tid = threadIdx.x;
    const int wave = tid >> 5, lane = tid & 31;
    const int wm = wave & 1, wn = wave >> 1;      // 2 x 4 waves -> 64x64 per wave
    const int lrow = lane & 15, lhi = lane >> 4;

    int arow[2], acol[2], brow[4], bcol[4];
    #pragma unroll
    for (int c = 0; c < 2; ++c) {
        int chunk = c * 256 + tid;                 // 512 chunks: 128 rows x 4
        arow[c] = chunk >> 2; acol[c] = (chunk & 3) * 8;
    }
    #pragma unroll
    for (int c = 0; c < 4; ++c) {
        int chunk = c * 256 + tid;                 // 1024 chunks: 256 rows x 4
        brow[c] = chunk >> 2; bcol[c] = (chunk & 3) * 8;
    }

    auto issue_tile = [&](int buf, int k0) {       // async DMA: HBM -> LDS
        #pragma unroll
        for (int c = 0; c < 2; ++c)
            async_ld16(&sm[buf][arow[c] * LDT + acol[c]],
                       A + (long)(bm * TM + arow[c]) * K + k0 + acol[c]);
        #pragma unroll
        for (int c = 0; c < 4; ++c)
            async_ld16(&sm[buf][(TM + brow[c]) * LDT + bcol[c]],
                       Bt + (long)(bn * TN + brow[c]) * K + k0 + bcol[c]);
    };

    v8f acc[4][4] = {};
    issue_tile(0, 0);
    wait_async0();
    __syncthreads();
    int buf = 0;
    for (int k0 = 0; k0 < K; k0 += TK) {
        const bool more = (k0 + TK) < K;
        if (more) issue_tile(buf ^ 1, k0 + TK);    // DMA next tile during WMMAs
        if (k0 + 2 * TK < K)                        // warm GL2 for tile after next
            __builtin_prefetch(Bt + (long)(bn * TN + brow[0]) * K + k0 + 2 * TK, 0, 1);

        Frag16 af[4];
        #pragma unroll
        for (int mi = 0; mi < 4; ++mi) {
            int row = wm * 64 + mi * 16 + lrow;
            int ko = lhi * 8;
            af[mi].h[0] = *(const v8bf*)(&sm[buf][row * LDT + ko]);
            af[mi].h[1] = *(const v8bf*)(&sm[buf][row * LDT + ko + 16]);
        }
        #pragma unroll
        for (int ni = 0; ni < 4; ++ni) {
            Frag16 bf;
            int row = TM + wn * 64 + ni * 16 + lrow;
            int kb = lhi * 8;
            bf.h[0] = *(const v8bf*)(&sm[buf][row * LDT + kb]);
            bf.h[1] = *(const v8bf*)(&sm[buf][row * LDT + kb + 16]);
            #pragma unroll
            for (int mi = 0; mi < 4; ++mi)
                acc[mi][ni] = wmma_bf16(af[mi], bf, acc[mi][ni]);
        }
        wait_async0();                              // this wave's DMA writes landed
        __syncthreads();                            // all waves' writes visible
        buf ^= 1;
    }
    #pragma unroll
    for (int mi = 0; mi < 4; ++mi)
        #pragma unroll
        for (int ni = 0; ni < 4; ++ni)
            #pragma unroll
            for (int r = 0; r < 8; ++r) {
                int row = bm * TM + wm * 64 + mi * 16 + lhi * 8 + r;
                int col = bn * TN + wn * 64 + ni * 16 + lrow;
                C[(long)row * N + col] = acc[mi][ni][r];
            }
}

// ---------------- RoPE: q -> bf16 [B,H,T,HD]; k roped in place (fp32) ----------------
__global__ __launch_bounds__(256) void k_rope(float* __restrict__ proj,
                                              __bf16* __restrict__ qb) {
    long idx = (long)blockIdx.x * 256 + threadIdx.x;   // (b,t,head,p)
    if (idx >= (long)BB * TT * (NH + HKV) * (HD / 2)) return;
    int p = (int)(idx & 63);
    long r1 = idx >> 6;
    int head = (int)(r1 % (NH + HKV));
    long bt = r1 / (NH + HKV);
    int t = (int)(bt % TT);
    float freq = __powf(10000.0f, -(float)p * (1.0f / 64.0f));
    float ang = (float)t * freq;
    float cs = __cosf(ang), sn = __sinf(ang);
    int col = (head < NH) ? head * HD : 2048 + (head - NH) * HD;
    float* base = proj + bt * NCAT + col + 2 * p;
    float a = base[0], b = base[1];
    float ra = a * cs - b * sn;
    float rb = a * sn + b * cs;
    if (head < NH) {
        int bidx = (int)(bt / TT);
        __bf16* qd = qb + (((long)(bidx * NH + head)) * TT + t) * HD + 2 * p;
        qd[0] = (__bf16)ra; qd[1] = (__bf16)rb;
    } else {
        base[0] = ra; base[1] = rb;
    }
}

// ---------------- block compression: ck/cv [B,HKV,NB,HD] fp32 ----------------
__global__ __launch_bounds__(128) void k_cblock(const float* __restrict__ proj,
                                                const float* __restrict__ cwa,
                                                const float* __restrict__ cwb,
                                                float* __restrict__ ck,
                                                float* __restrict__ cv) {
    int n = blockIdx.x % NB;
    int rest = blockIdx.x / NB;
    int hkv = rest % HKV, b = rest / HKV;
    int tid = threadIdx.x;
    __shared__ float sa[RR], sb[RR], w[RR];
    const float* kb = proj + ((long)(b * TT + n * RR)) * NCAT + 2048 + hkv * HD;
    const float* vb = proj + ((long)(b * TT + n * RR)) * NCAT + 2560 + hkv * HD;
    if (tid < RR) {
        float da = 0.f, db = 0.f;
        for (int d = 0; d < HD; ++d) {
            float kv = kb[(long)tid * NCAT + d];
            da += kv * cwa[d]; db += kv * cwb[d];
        }
        sa[tid] = da; sb[tid] = db;
    }
    __syncthreads();
    if (tid == 0) {
        float ma = -3.0e38f, mb = -3.0e38f;
        for (int r = 0; r < RR; ++r) { ma = fmaxf(ma, sa[r]); mb = fmaxf(mb, sb[r]); }
        float ea[RR], eb[RR], sua = 0.f, sub = 0.f;
        for (int r = 0; r < RR; ++r) {
            ea[r] = __expf(sa[r] - ma); sua += ea[r];
            eb[r] = __expf(sb[r] - mb); sub += eb[r];
        }
        for (int r = 0; r < RR; ++r) w[r] = 0.5f * (ea[r] / sua + eb[r] / sub);
    }
    __syncthreads();
    int d = tid;   // HD == 128 == blockDim
    float ak = 0.f, av = 0.f;
    for (int r = 0; r < RR; ++r) {
        ak += w[r] * kb[(long)r * NCAT + d];
        av += w[r] * vb[(long)r * NCAT + d];
    }
    long o = (((long)(b * HKV + hkv)) * NB + n) * HD + d;
    ck[o] = ak; cv[o] = av;
}

// ---------------- ki_mean[b][d] = mean_t proj[...,3200+d] ----------------
__global__ __launch_bounds__(256) void k_kimean(const float* __restrict__ proj,
                                                float* __restrict__ kim) {
    int d = blockIdx.x % HD, b = blockIdx.x / HD;
    float s = 0.f;
    for (int t = threadIdx.x; t < TT; t += 256)
        s += proj[((long)(b * TT + t)) * NCAT + 3200 + d];
    __shared__ float red[256];
    red[threadIdx.x] = s; __syncthreads();
    for (int o = 128; o > 0; o >>= 1) {
        if (threadIdx.x < o) red[threadIdx.x] += red[threadIdx.x + o];
        __syncthreads();
    }
    if (threadIdx.x == 0) kim[b * HD + d] = red[0] * (1.0f / (float)TT);
}

// ---------------- scores (t < NB) + top-64 selection ----------------
__global__ __launch_bounds__(256) void k_topk(const float* __restrict__ proj,
                                              const float* __restrict__ kim,
                                              int* __restrict__ sel) {
    int b = blockIdx.x, t = threadIdx.x;   // NB == 256 == blockDim
    __shared__ float sc[NB], rv[NB];
    __shared__ int   ri[NB];
    const float* qi = proj + ((long)(b * TT + t)) * NCAT + 3072;
    float s = 0.f;
    for (int d = 0; d < HD; ++d) s += qi[d] * kim[b * HD + d];
    sc[t] = s; __syncthreads();
    for (int i = 0; i < TOPK; ++i) {
        rv[t] = sc[t]; ri[t] = t; __syncthreads();
        for (int o = 128; o > 0; o >>= 1) {
            if (t < o && rv[t + o] > rv[t]) { rv[t] = rv[t + o]; ri[t] = ri[t + o]; }
            __syncthreads();
        }
        if (t == 0) { sel[b * TOPK + i] = ri[0]; sc[ri[0]] = -3.0e38f; }
        __syncthreads();
    }
}

// ---------------- gather selected blocks -> bf16 (ckb row-major, cvT transposed) ----------------
__global__ __launch_bounds__(256) void k_gather(const float* __restrict__ ck,
                                                const float* __restrict__ cv,
                                                const int* __restrict__ sel,
                                                __bf16* __restrict__ ckb,
                                                __bf16* __restrict__ cvT) {
    long idx = (long)blockIdx.x * 256 + threadIdx.x;
    if (idx >= (long)BB * HKV * TOPK * HD) return;
    int d = (int)(idx % HD);
    long r = idx / HD;
    int s = (int)(r % TOPK);
    long r2 = r / TOPK;
    int hkv = (int)(r2 % HKV), b = (int)(r2 / HKV);
    int n = sel[b * TOPK + s];
    long src = (((long)(b * HKV + hkv)) * NB + n) * HD + d;
    ckb[(((long)(b * HKV + hkv)) * TOPK + s) * HD + d] = (__bf16)ck[src];
    cvT[(((long)(b * HKV + hkv)) * HD + d) * TOPK + s] = (__bf16)cv[src];
}

// ---------------- WMMA attention: 16 queries x 64 keys x 128 dims per wave ----------------
__global__ __launch_bounds__(256) void k_attn(const __bf16* __restrict__ qb,
                                              const __bf16* __restrict__ ckb,
                                              const __bf16* __restrict__ cvT,
                                              __bf16* __restrict__ out_att) {
    const int tid = threadIdx.x, wave = tid >> 5, lane = tid & 31;
    const int lrow = lane & 15, lhi = lane >> 4;
    int tchunk = blockIdx.x % (TT / 128);
    int bh = blockIdx.x / (TT / 128);
    int h = bh % NH, b = bh / NH;
    int hkv = h / NREP;
    int t0 = tchunk * 128 + wave * 16;

    const __bf16* qp = qb  + (((long)(b * NH + h)) * TT + t0) * HD;
    const __bf16* kp = ckb + ((long)(b * HKV + hkv)) * TOPK * HD;
    const __bf16* vp = cvT + ((long)(b * HKV + hkv)) * HD * TOPK;

    // logits[16 x 64] = q[16x128] @ ck^T
    v8f lacc[4] = {};
    #pragma unroll
    for (int kk = 0; kk < 4; ++kk) {
        Frag16 af;
        af.h[0] = *(const v8bf*)(qp + (long)lrow * HD + kk * 32 + lhi * 8);
        af.h[1] = *(const v8bf*)(qp + (long)lrow * HD + kk * 32 + lhi * 8 + 16);
        #pragma unroll
        for (int ns = 0; ns < 4; ++ns) {
            Frag16 bf;
            int key = ns * 16 + lrow;
            bf.h[0] = *(const v8bf*)(kp + (long)key * HD + kk * 32 + lhi * 8);
            bf.h[1] = *(const v8bf*)(kp + (long)key * HD + kk * 32 + lhi * 8 + 16);
            lacc[ns] = wmma_bf16(af, bf, lacc[ns]);
        }
    }
    // rowwise softmax in C-fragment layout (row = r + 8*lhi, 16 cols/frag across lanes)
    const float scale = 0.08838834764831845f;   // 1/sqrt(128)
    __shared__ __bf16 attS[8][16][72];          // per wave, 8-elem pad
    #pragma unroll
    for (int r = 0; r < 8; ++r) {
        float mx = -3.0e38f;
        #pragma unroll
        for (int ns = 0; ns < 4; ++ns) mx = fmaxf(mx, lacc[ns][r]);
        for (int m = 1; m < 16; m <<= 1) mx = fmaxf(mx, __shfl_xor(mx, m, 32));
        float e[4], sum = 0.f;
        #pragma unroll
        for (int ns = 0; ns < 4; ++ns) {
            e[ns] = __expf((lacc[ns][r] - mx) * scale);
            sum += e[ns];
        }
        for (int m = 1; m < 16; m <<= 1) sum += __shfl_xor(sum, m, 32);
        float inv = 1.0f / sum;
        int rowm = r + lhi * 8;
        #pragma unroll
        for (int ns = 0; ns < 4; ++ns)
            attS[wave][rowm][ns * 16 + lrow] = (__bf16)(e[ns] * inv);
    }
    __syncthreads();
    // out[16 x 128] = att[16x64] @ cv  (cv stored transposed [d][s])
    v8f oacc[8] = {};
    #pragma unroll
    for (int kk = 0; kk < 2; ++kk) {
        Frag16 af;
        af.h[0] = *(const v8bf*)(&attS[wave][lrow][kk * 32 + lhi * 8]);
        af.h[1] = *(const v8bf*)(&attS[wave][lrow][kk * 32 + lhi * 8 + 16]);
        #pragma unroll
        for (int nf = 0; nf < 8; ++nf) {
            Frag16 bf;
            int d = nf * 16 + lrow;
            bf.h[0] = *(const v8bf*)(vp + (long)d * TOPK + kk * 32 + lhi * 8);
            bf.h[1] = *(const v8bf*)(vp + (long)d * TOPK + kk * 32 + lhi * 8 + 16);
            oacc[nf] = wmma_bf16(af, bf, oacc[nf]);
        }
    }
    #pragma unroll
    for (int nf = 0; nf < 8; ++nf)
        #pragma unroll
        for (int r = 0; r < 8; ++r) {
            int t = t0 + r + lhi * 8;
            int d = nf * 16 + lrow;
            out_att[((long)b * TT + t) * DIM + h * HD + d] = (__bf16)oacc[nf][r];
        }
}

// ---------------- host launch ----------------
extern "C" void kernel_launch(void* const* d_in, const int* in_sizes, int n_in,
                              void* d_out, int out_size, void* d_ws, size_t ws_size,
                              hipStream_t stream) {
    const float* x   = (const float*)d_in[0];
    const float* wq  = (const float*)d_in[1];
    const float* wk  = (const float*)d_in[2];
    const float* wv  = (const float*)d_in[3];
    const float* wo  = (const float*)d_in[4];
    const float* wiq = (const float*)d_in[5];
    const float* wik = (const float*)d_in[6];
    const float* cwa = (const float*)d_in[7];
    const float* cwb = (const float*)d_in[8];
    float* out = (float*)d_out;

    char* p = (char*)d_ws;
    auto take = [&](size_t bytes) { char* q = p; p += (bytes + 255) & ~size_t(255); return q; };
    __bf16* xb     = (__bf16*)take((size_t)MTOT * DIM * 2);
    __bf16* wcatT  = (__bf16*)take((size_t)NCAT * DIM * 2);
    __bf16* woT    = (__bf16*)take((size_t)DIM * DIM * 2);
    float*  proj   = (float*) take((size_t)MTOT * NCAT * 4);
    __bf16* qb     = (__bf16*)take((size_t)BB * NH * TT * HD * 2);
    float*  ck     = (float*) take((size_t)BB * HKV * NB * HD * 4);
    float*  cv     = (float*) take((size_t)BB * HKV * NB * HD * 4);
    float*  kim    = (float*) take((size_t)BB * HD * 4);
    int*    sel    = (int*)   take((size_t)BB * TOPK * 4);
    __bf16* ckb    = (__bf16*)take((size_t)BB * HKV * TOPK * HD * 2);
    __bf16* cvT    = (__bf16*)take((size_t)BB * HKV * HD * TOPK * 2);
    __bf16* attout = (__bf16*)take((size_t)MTOT * DIM * 2);

    // 1) convert x; pack/transpose weights (one-time, K-contiguous layouts)
    {
        long n = (long)MTOT * DIM;
        k_cvt_bf16<<<dim3((unsigned)((n + 255) / 256)), dim3(256), 0, stream>>>(x, xb, n);
    }
    {
        long n = (long)DIM * DIM;
        k_transpose_wo<<<dim3((unsigned)((n + 255) / 256)), dim3(256), 0, stream>>>(wo, woT);
    }
    {
        long n = (long)DIM * NCAT;
        k_pack_wcatT<<<dim3((unsigned)((n + 255) / 256)), dim3(256), 0, stream>>>(
            wq, wk, wv, wiq, wik, wcatT);
    }
    // 2) fused projection GEMM: proj = xb @ wcatT^T   (8192 x 3328 x 2048)
    k_gemm_bf16<<<dim3(NCAT / TN, MTOT / TM), dim3(256), 0, stream>>>(
        xb, wcatT, proj, MTOT, NCAT, DIM);
    // 3) RoPE
    {
        long n = (long)BB * TT * (NH + HKV) * (HD / 2);
        k_rope<<<dim3((unsigned)((n + 255) / 256)), dim3(256), 0, stream>>>(proj, qb);
    }
    // 4) block compression, 5) ki mean, 6) scores + top-k, 7) gather
    k_cblock<<<dim3(BB * HKV * NB), dim3(128), 0, stream>>>(proj, cwa, cwb, ck, cv);
    k_kimean<<<dim3(BB * HD), dim3(256), 0, stream>>>(proj, kim);
    k_topk<<<dim3(BB), dim3(256), 0, stream>>>(proj, kim, sel);
    {
        long n = (long)BB * HKV * TOPK * HD;
        k_gather<<<dim3((unsigned)((n + 255) / 256)), dim3(256), 0, stream>>>(
            ck, cv, sel, ckb, cvT);
    }
    // 8) WMMA attention
    k_attn<<<dim3(BB * NH * (TT / 128)), dim3(256), 0, stream>>>(qb, ckb, cvT, attout);
    // 9) output projection GEMM -> d_out
    k_gemm_bf16<<<dim3(DIM / TN, MTOT / TM), dim3(256), 0, stream>>>(
        attout, woT, out, MTOT, DIM, DIM);
    (void)in_sizes; (void)n_in; (void)out_size; (void)ws_size;
}
